// GNN_PBE_HeteroTransformerConv_22909355557013
// MI455X (gfx1250) — matrix operand order repaired and checked by reading, not compiled, not run on hardware
//
#include <hip/hip_runtime.h>
#include <hip/hip_bf16.h>
#include <stdint.h>

// ---------------------------------------------------------------------------
// GNN power-flow forward pass for MI455X (gfx1250).
// Dense GEMMs: f32 -> bf16 fragment packing, then V_WMMA_F32_16X16X32_BF16
// with 4 output tiles (16x64) per wave and A-fragment reuse.
// Irregular graph ops (segment softmax, scatters, PF residual): VALU + atomics.
// Workspace requirement: ~770 MB.
// ---------------------------------------------------------------------------

#define NB   30000
#define NG   6000
#define EBB  120000
#define HID  128
#define HC   512
#define HEADS 4

typedef __attribute__((ext_vector_type(16))) __bf16 v16bf;
typedef __attribute__((ext_vector_type(8)))  float  v8f;

__device__ __forceinline__ unsigned bfbits(float f) {
  unsigned u = __builtin_bit_cast(unsigned, f);
  u += 0x7FFFu + ((u >> 16) & 1u);               // round-to-nearest-even
  return u >> 16;
}
__device__ __forceinline__ unsigned pack2(float lo, float hi) {
  return bfbits(lo) | (bfbits(hi) << 16);
}

__device__ __forceinline__ float lrelu(float x) { return x > 0.f ? x : 0.01f * x; }

// monotone float<->uint mapping for atomicMax on floats
__device__ __forceinline__ unsigned flipf(float f) {
  unsigned u = __builtin_bit_cast(unsigned, f);
  return (u & 0x80000000u) ? ~u : (u | 0x80000000u);
}
__device__ __forceinline__ float unflipf(unsigned u) {
  unsigned v = (u & 0x80000000u) ? (u & 0x7FFFFFFFu) : ~u;
  return __builtin_bit_cast(float, v);
}

// ---------------------------------------------------------------------------
// Fragment packing: one thread per 32-byte lane-chunk.
// Chunk id t = ((tile * KT) + kt) * 32 + lane; KT = K/32.
// A chunk (16x32 tile): lane&15 = row-in-tile; lane>>4 selects K halves
//   halfwords 0..7  = A[row][kt*32 + (lane>>4)*8 + 0..7]
//   halfwords 8..15 = A[row][kt*32 + (lane>>4)*8 + 16..23]
// B chunk (32x16 tile): lane&15 = col-in-tile;
//   halfwords 0..7  = B[kt*32 + (lane>>4)*8 + 0..7][col]
//   halfwords 8..15 = B[kt*32 + (lane>>4)*8 + 16..23][col]
// ---------------------------------------------------------------------------
__global__ void pack_a_kernel(const float* __restrict__ A, unsigned* __restrict__ Ap,
                              int M, int K) {
  const int KT = K >> 5;
  const int total = (M >> 4) * KT * 32;
  int t = blockIdx.x * blockDim.x + threadIdx.x;
  if (t >= total) return;
  const int lane = t & 31;
  const int kt = (t >> 5) % KT;
  const int mtile = (t >> 5) / KT;
  const int row = mtile * 16 + (lane & 15);
  const float* Ar = A + (size_t)row * K + kt * 32 + (lane >> 4) * 8;
  unsigned w[8];
#pragma unroll
  for (int j = 0; j < 4; ++j) w[j]     = pack2(Ar[2*j],      Ar[2*j + 1]);
#pragma unroll
  for (int j = 0; j < 4; ++j) w[j + 4] = pack2(Ar[16 + 2*j], Ar[16 + 2*j + 1]);
  uint4* o = (uint4*)(Ap + (size_t)t * 8);
  o[0] = make_uint4(w[0], w[1], w[2], w[3]);
  o[1] = make_uint4(w[4], w[5], w[6], w[7]);
}

__global__ void pack_b_kernel(const float* __restrict__ B, unsigned* __restrict__ Bp,
                              int N, int K) {
  const int KT = K >> 5;
  const int total = (N >> 4) * KT * 32;
  int t = blockIdx.x * blockDim.x + threadIdx.x;
  if (t >= total) return;
  const int lane = t & 31;
  const int kt = (t >> 5) % KT;
  const int ntile = (t >> 5) / KT;
  const int col = ntile * 16 + (lane & 15);
  const int kb = kt * 32 + (lane >> 4) * 8;
  unsigned w[8];
#pragma unroll
  for (int j = 0; j < 8; ++j)
    w[j] = pack2(B[(size_t)(kb + 2*j + (j >= 4 ? 8 : 0)) * N + col], B[(size_t)(kb + 2*j + 1 + (j >= 4 ? 8 : 0)) * N + col]);
  uint4* o = (uint4*)(Bp + (size_t)t * 8);
  o[0] = make_uint4(w[0], w[1], w[2], w[3]);
  o[1] = make_uint4(w[4], w[5], w[6], w[7]);
}

// ---------------------------------------------------------------------------
// Packed WMMA GEMM: C[M,N] = A @ B + bias.
// Wave computes 16x64 strip: A fragment loaded once, 4 B fragments, 4 WMMAs.
// 4 waves (128 threads) per block.
// ---------------------------------------------------------------------------
__global__ void wmma_gemm_packed_kernel(const v16bf* __restrict__ Ap,
                                        const v16bf* __restrict__ Bp,
                                        const float* __restrict__ bias,
                                        float* __restrict__ C,
                                        int M, int N, int K) {
  const int lane = threadIdx.x & 31;
  const int wave = threadIdx.x >> 5;
  const int mtile = blockIdx.y * 4 + wave;
  if (mtile * 16 >= M) return;
  const int KT = K >> 5;
  const int nt0 = blockIdx.x * 4;
  const size_t aBase = (size_t)mtile * KT * 32 + lane;
  const size_t bStride = (size_t)KT * 32;            // per column-tile
  const size_t bBase = (size_t)nt0 * bStride + lane;
  v8f acc0 = {}, acc1 = {}, acc2 = {}, acc3 = {};
  for (int kt = 0; kt < KT; ++kt) {
    if (kt + 1 < KT) __builtin_prefetch(&Ap[aBase + (size_t)(kt + 1) * 32], 0, 0);
    v16bf a  = Ap[aBase + (size_t)kt * 32];
    v16bf b0 = Bp[bBase + (size_t)kt * 32];
    v16bf b1 = Bp[bBase + bStride + (size_t)kt * 32];
    v16bf b2 = Bp[bBase + 2 * bStride + (size_t)kt * 32];
    v16bf b3 = Bp[bBase + 3 * bStride + (size_t)kt * 32];
    acc0 = __builtin_amdgcn_wmma_f32_16x16x32_bf16(false, a, false, b0, (short)0, acc0, false, false);
    acc1 = __builtin_amdgcn_wmma_f32_16x16x32_bf16(false, a, false, b1, (short)0, acc1, false, false);
    acc2 = __builtin_amdgcn_wmma_f32_16x16x32_bf16(false, a, false, b2, (short)0, acc2, false, false);
    acc3 = __builtin_amdgcn_wmma_f32_16x16x32_bf16(false, a, false, b3, (short)0, acc3, false, false);
  }
  const int r = lane & 15, hi = lane >> 4;
  const int row0 = mtile * 16 + hi * 8;
#pragma unroll
  for (int nti = 0; nti < 4; ++nti) {
    const int col = (nt0 + nti) * 16 + r;
    const float bz = bias ? bias[col] : 0.f;
    const v8f acc = (nti == 0) ? acc0 : (nti == 1) ? acc1 : (nti == 2) ? acc2 : acc3;
#pragma unroll
    for (int i = 0; i < 8; ++i)
      C[(size_t)(row0 + i) * N + col] = acc[i] + bz;
  }
}

// ---------------------------------------------------------------------------
// _proj: Out = LN(leaky(X@W1+b1) @ W2 + b2) * g + be       (HID = 128 wide)
// ---------------------------------------------------------------------------
__global__ void proj_kernel(const float* __restrict__ X, int din,
                            const float* __restrict__ W1, const float* __restrict__ b1,
                            const float* __restrict__ W2, const float* __restrict__ b2,
                            const float* __restrict__ g,  const float* __restrict__ be,
                            float* __restrict__ Out, int rows) {
  int row = blockIdx.x;
  if (row >= rows) return;
  int j = threadIdx.x;
  __shared__ float h1[128];
  __shared__ float red[128];
  const float* xr = X + (size_t)row * din;
  float s = b1[j];
  for (int d = 0; d < din; ++d) s += xr[d] * W1[d * 128 + j];
  h1[j] = lrelu(s);
  __syncthreads();
  float s2 = b2[j];
  for (int k = 0; k < 128; ++k) s2 += h1[k] * W2[k * 128 + j];
  red[j] = s2; __syncthreads();
  for (int off = 64; off > 0; off >>= 1) { if (j < off) red[j] += red[j+off]; __syncthreads(); }
  float mean = red[0] * (1.f/128.f); __syncthreads();
  float dx = s2 - mean;
  red[j] = dx * dx; __syncthreads();
  for (int off = 64; off > 0; off >>= 1) { if (j < off) red[j] += red[j+off]; __syncthreads(); }
  float var = red[0] * (1.f/128.f);
  Out[(size_t)row * 128 + j] = dx * rsqrtf(var + 1e-5f) * g[j] + be[j];
}

// LN over `width` + leaky, optional residual add. blockDim == width.
__global__ void ln_leaky_kernel(const float* __restrict__ In,
                                const float* __restrict__ g, const float* __restrict__ b,
                                float* __restrict__ Out, int width, int rows, int addres) {
  int row = blockIdx.x;
  if (row >= rows) return;
  int j = threadIdx.x;
  extern __shared__ float red[];
  size_t o = (size_t)row * width + j;
  float x = In[o];
  red[j] = x; __syncthreads();
  for (int off = width >> 1; off > 0; off >>= 1) { if (j < off) red[j] += red[j+off]; __syncthreads(); }
  float mean = red[0] / (float)width; __syncthreads();
  float dx = x - mean;
  red[j] = dx * dx; __syncthreads();
  for (int off = width >> 1; off > 0; off >>= 1) { if (j < off) red[j] += red[j+off]; __syncthreads(); }
  float var = red[0] / (float)width;
  float y = lrelu(dx * rsqrtf(var + 1e-5f) * g[j] + b[j]);
  if (addres) Out[o] += y; else Out[o] = y;
}

// ---------------------------------------------------------------------------
// attention: one wave per (edge, head)
// ---------------------------------------------------------------------------
__global__ void attn_score_kernel(const float* __restrict__ q, const float* __restrict__ k,
                                  const float* __restrict__ ee,
                                  const int* __restrict__ src, const int* __restrict__ dst,
                                  int E, float* __restrict__ a, unsigned* __restrict__ amax) {
  int unit = blockIdx.x * (blockDim.x >> 5) + (threadIdx.x >> 5);
  if (unit >= E * HEADS) return;
  int e = unit >> 2, h = unit & 3, lane = threadIdx.x & 31;
  int s = src[e], d = dst[e];
  const float* qp = q + (size_t)d * HC + h * HID;
  const float* kp = k + (size_t)s * HC + h * HID;
  float part = 0.f;
  if (ee) {
    const float* ep = ee + (size_t)e * HC + h * HID;
    for (int i = lane; i < HID; i += 32) part += qp[i] * (kp[i] + ep[i]);
  } else {
    for (int i = lane; i < HID; i += 32) part += qp[i] * kp[i];
  }
  for (int off = 16; off > 0; off >>= 1) part += __shfl_down(part, off, 32);
  if (lane == 0) {
    float sc = part * 0.08838834764831845f;    // 1/sqrt(128)
    a[(size_t)e * HEADS + h] = sc;
    atomicMax(&amax[(size_t)d * HEADS + h], flipf(sc));
  }
}

__global__ void attn_exp_kernel(float* __restrict__ a, const unsigned* __restrict__ amax,
                                const int* __restrict__ dst, int E, float* __restrict__ den) {
  int t = blockIdx.x * blockDim.x + threadIdx.x;
  if (t >= E * HEADS) return;
  int e = t >> 2, h = t & 3;
  int d = dst[e];
  float am = unflipf(amax[(size_t)d * HEADS + h]);
  if (!__builtin_isfinite(am)) am = 0.f;
  float ex = expf(a[t] - am);
  a[t] = ex;
  atomicAdd(&den[(size_t)d * HEADS + h], ex);
}

__global__ void attn_accum_kernel(const float* __restrict__ ex, const float* __restrict__ den,
                                  const float* __restrict__ v, const float* __restrict__ ee,
                                  const int* __restrict__ src, const int* __restrict__ dst,
                                  int E, float* __restrict__ out) {
  int unit = blockIdx.x * (blockDim.x >> 5) + (threadIdx.x >> 5);
  if (unit >= E * HEADS) return;
  int e = unit >> 2, h = unit & 3, lane = threadIdx.x & 31;
  int s = src[e], d = dst[e];
  float w = ex[(size_t)e * HEADS + h] / (den[(size_t)d * HEADS + h] + 1e-16f);
  const float* vp = v + (size_t)s * HC + h * HID;
  float* op = out + (size_t)d * HC + h * HID;
  if (ee) {
    const float* ep = ee + (size_t)e * HC + h * HID;
    for (int i = lane; i < HID; i += 32) atomicAdd(&op[i], w * (vp[i] + ep[i]));
  } else {
    for (int i = lane; i < HID; i += 32) atomicAdd(&op[i], w * vp[i]);
  }
}

// beta gate: beta = sigmoid([out, r, out-r] @ Wbeta); m (+)= beta*r + (1-beta)*out
__global__ void beta_blend_kernel(const float* __restrict__ outm, const float* __restrict__ r,
                                  const float* __restrict__ Wb, float* __restrict__ m,
                                  int rows, int accumulate) {
  int row = blockIdx.x;
  if (row >= rows) return;
  int j = threadIdx.x;  // 256
  __shared__ float red[256];
  size_t base = (size_t)row * HC;
  float part = 0.f;
#pragma unroll
  for (int t = 0; t < 2; ++t) {
    int c = j + t * 256;
    float o = outm[base + c], rr = r[base + c];
    part += o * Wb[c] + rr * Wb[HC + c] + (o - rr) * Wb[2 * HC + c];
  }
  red[j] = part; __syncthreads();
  for (int off = 128; off > 0; off >>= 1) { if (j < off) red[j] += red[j+off]; __syncthreads(); }
  float beta = 1.f / (1.f + expf(-red[0]));
#pragma unroll
  for (int t = 0; t < 2; ++t) {
    int c = j + t * 256;
    float o = outm[base + c], rr = r[base + c];
    float val = beta * rr + (1.f - beta) * o;
    if (accumulate) m[base + c] += val; else m[base + c] = val;
  }
}

// out-mlp head: outt = mask ? hid@W2+b2 : fixed[:, :dout]
__global__ void out_head_kernel(const float* __restrict__ hid, const float* __restrict__ W2,
                                const float* __restrict__ b2, const uint8_t* __restrict__ mask,
                                const float* __restrict__ fixedv, int fixld,
                                float* __restrict__ outt, int rows, int dout) {
  int t = blockIdx.x * blockDim.x + threadIdx.x;
  if (t >= rows * dout) return;
  int row = t / dout, o = t - row * dout;
  float s = b2[o];
  const float* hp = hid + (size_t)row * 128;
  for (int k = 0; k < 128; ++k) s += hp[k] * W2[k * dout + o];
  outt[t] = mask[t] ? s : fixedv[(size_t)row * fixld + o];
}

__global__ void agg_kernel(const float* __restrict__ gen_t, const int* __restrict__ gsrc,
                           const int* __restrict__ gdst, int ng, float* __restrict__ agg) {
  int t = blockIdx.x * blockDim.x + threadIdx.x;
  if (t >= ng * 2) return;
  int gi = t >> 1, c = t & 1;
  atomicAdd(&agg[(size_t)gdst[gi] * 2 + c], gen_t[(size_t)gsrc[gi] * 2 + c]);
}

__global__ void vrvi_kernel(const float* __restrict__ bus_t, float* __restrict__ vri, int nb) {
  int d = blockIdx.x * blockDim.x + threadIdx.x;
  if (d >= nb) return;
  float vm = bus_t[(size_t)d * 4 + 2], va = bus_t[(size_t)d * 4 + 3];
  vri[(size_t)d * 2]     = vm * cosf(va);
  vri[(size_t)d * 2 + 1] = vm * sinf(va);
}

__global__ void pf_edge_kernel(const float* __restrict__ ea, const int* __restrict__ src,
                               const int* __restrict__ dst, const float* __restrict__ vri,
                               float* __restrict__ irii, int E) {
  int e = blockIdx.x * blockDim.x + threadIdx.x;
  if (e >= E) return;
  float G = ea[(size_t)e * 2], B = ea[(size_t)e * 2 + 1];
  int s = src[e], d = dst[e];
  float vr = vri[(size_t)s * 2], vi = vri[(size_t)s * 2 + 1];
  atomicAdd(&irii[(size_t)d * 2],     G * vr - B * vi);
  atomicAdd(&irii[(size_t)d * 2 + 1], G * vi + B * vr);
}

__global__ void pf_res_kernel(const float* __restrict__ bus_t, const float* __restrict__ agg,
                              const float* __restrict__ vri, const float* __restrict__ irii,
                              float* __restrict__ res, int nb) {
  int d = blockIdx.x * blockDim.x + threadIdx.x;
  if (d >= nb) return;
  float vr = vri[(size_t)d * 2], vi = vri[(size_t)d * 2 + 1];
  float ir = irii[(size_t)d * 2], ii = irii[(size_t)d * 2 + 1];
  float p = vr * ir + vi * ii;
  float q = vi * ir - vr * ii;
  res[(size_t)d * 2]     = p - (agg[(size_t)d * 2]     - bus_t[(size_t)d * 4]);
  res[(size_t)d * 2 + 1] = q - (agg[(size_t)d * 2 + 1] - bus_t[(size_t)d * 4 + 1]);
}

__global__ void phys_kernel(const float* __restrict__ res, const float* __restrict__ W,
                            const float* __restrict__ b, float* __restrict__ h, int nb) {
  int t = blockIdx.x * blockDim.x + threadIdx.x;
  if (t >= nb * HC) return;
  int d = t >> 9, c = t & (HC - 1);
  float v = res[(size_t)d * 2] * W[c] + res[(size_t)d * 2 + 1] * W[HC + c] + b[c];
  h[t] += lrelu(v);
}

__global__ void final_out_kernel(const float* __restrict__ bus_f, const float* __restrict__ agg,
                                 float* __restrict__ out, int nb) {
  int t = blockIdx.x * blockDim.x + threadIdx.x;
  if (t >= nb * 6) return;
  int d = t / 6, c = t - d * 6;
  out[t] = (c < 4) ? bus_f[(size_t)d * 4 + c] : agg[(size_t)d * 2 + (c - 4)];
}

// ---------------------------------------------------------------------------
extern "C" void kernel_launch(void* const* d_in, const int* in_sizes, int n_in,
                              void* d_out, int out_size, void* d_ws, size_t ws_size,
                              hipStream_t stream) {
  (void)in_sizes; (void)n_in; (void)out_size; (void)ws_size;
  auto P = [&](int i) { return (const float*)d_in[i]; };

  // ---- input map (setup_inputs insertion order; params flattened with
  //      JAX-style sorted dict keys) ----
  const float* x_bus = P(0);
  const float* x_gen = P(1);
  const float* ea_bb = P(2);
  // params base 3: layers(96) mlp_bus(6) mlp_gen(6) phys(2) proj_bus(6) proj_edge(6) proj_gen(6)
  const int MLPB = 99, MLPG = 105, PHYS = 111, PRB = 113, PRE = 119, PRG = 125;
  const int* eibb = (const int*)d_in[131];
  const int* eigb = (const int*)d_in[132];
  const int* eibg = (const int*)d_in[133];
  const uint8_t* mask_bus = (const uint8_t*)d_in[134];
  const uint8_t* mask_gen = (const uint8_t*)d_in[135];
  const int *bb_s = eibb, *bb_d = eibb + EBB;
  const int *gb_s = eigb, *gb_d = eigb + NG;
  const int *bg_s = eibg, *bg_d = eibg + NG;

  // ---- workspace layout (floats; all sizes multiples of 8 -> 32B aligned) ----
  float* w = (float*)d_ws;
  size_t off = 0;
  auto alloc = [&](size_t n) { float* p = w + off; off += n; return p; };
  float* h_bus = alloc((size_t)NB * HC);
  float* h_gen = alloc((size_t)NG * HC);
  float* e_emb = alloc((size_t)EBB * HID);
  float* qb    = alloc((size_t)NB * HC);   // q, later reused as message accumulator
  float* kb    = alloc((size_t)NB * HC);
  float* vb    = alloc((size_t)NB * HC);
  float* sk    = alloc((size_t)NB * HC);
  float* mb    = alloc((size_t)NB * HC);
  float* mg    = alloc((size_t)NG * HC);
  float* eeb   = alloc((size_t)EBB * HC);
  float* scb   = alloc((size_t)EBB * HEADS);
  float* amaxb = alloc((size_t)NB * HEADS);
  float* denb  = alloc((size_t)NB * HEADS);
  float* hidb  = alloc((size_t)NB * HID);
  float* h0b   = alloc((size_t)NB * HID);
  float* h0g   = alloc((size_t)NG * HID);
  float* bus_t = alloc((size_t)NB * 4);
  float* gen_t = alloc((size_t)NG * 2);
  float* aggb  = alloc((size_t)NB * 2);
  float* vrib  = alloc((size_t)NB * 2);
  float* irii  = alloc((size_t)NB * 2);
  float* resb  = alloc((size_t)NB * 2);
  unsigned* apack = (unsigned*)alloc((size_t)NB * HC / 2);   // max M*K bf16 = 15.36M
  unsigned* bpack = (unsigned*)alloc((size_t)HC * HC / 2);   // max K*N bf16 = 262144

  auto zero = [&](float* p, size_t n) { hipMemsetAsync(p, 0, n * sizeof(float), stream); };

  auto gemm = [&](const float* A, const float* B, const float* bias, float* C,
                  int M, int N, int K) {
    const int KT = K >> 5;
    const int ca = (M >> 4) * KT * 32;
    const int cb = (N >> 4) * KT * 32;
    pack_a_kernel<<<(ca + 255) / 256, 256, 0, stream>>>(A, apack, M, K);
    pack_b_kernel<<<(cb + 255) / 256, 256, 0, stream>>>(B, bpack, N, K);
    dim3 grid(N / 64, (M / 16 + 3) / 4);
    wmma_gemm_packed_kernel<<<grid, 128, 0, stream>>>(
        (const v16bf*)apack, (const v16bf*)bpack, bias, C, M, N, K);
  };

  // ---- transformer conv ----
  auto tconv = [&](const float* xsrc, int nsrc, const float* xdst, int ndst, int K,
                   int pidx, bool edge, const int* src, const int* dst, int E,
                   float* mdst, int accumulate) {
    int o = edge ? 1 : 0;
    const float* Wbeta = P(pidx + 0);
    const float* We    = edge ? P(pidx + 1) : nullptr;
    const float* Wk = P(pidx + 1 + o), *Wq = P(pidx + 2 + o);
    const float* Wsk = P(pidx + 3 + o), *Wv = P(pidx + 4 + o);
    const float* bk = P(pidx + 5 + o), *bq = P(pidx + 6 + o);
    const float* bsk = P(pidx + 7 + o), *bv = P(pidx + 8 + o);
    gemm(xdst, Wq,  bq,  qb, ndst, HC, K);
    gemm(xsrc, Wk,  bk,  kb, nsrc, HC, K);
    gemm(xsrc, Wv,  bv,  vb, nsrc, HC, K);
    gemm(xdst, Wsk, bsk, sk, ndst, HC, K);
    if (edge) gemm(e_emb, We, nullptr, eeb, EBB, HC, HID);
    zero(amaxb, (size_t)ndst * HEADS);   // 0u decodes to NaN -> treated as 0
    zero(denb,  (size_t)ndst * HEADS);
    int units = E * HEADS;
    attn_score_kernel<<<(units + 7) / 8, 256, 0, stream>>>(
        qb, kb, edge ? eeb : nullptr, src, dst, E, scb, (unsigned*)amaxb);
    attn_exp_kernel<<<(units + 255) / 256, 256, 0, stream>>>(
        scb, (const unsigned*)amaxb, dst, E, denb);
    zero(qb, (size_t)ndst * HC);         // reuse q buffer as message accumulator
    attn_accum_kernel<<<(units + 7) / 8, 256, 0, stream>>>(
        scb, denb, vb, edge ? eeb : nullptr, src, dst, E, qb);
    beta_blend_kernel<<<ndst, 256, 0, stream>>>(qb, sk, Wbeta, mdst, ndst, accumulate);
  };

  // ---- output MLP with mask ----
  auto outmlp = [&](const float* h, int n, int base, const uint8_t* mask,
                    const float* fixedv, int fixld, int dout, float* outt) {
    gemm(h, P(base + 0), P(base + 2), hidb, n, HID, HC);           // @W1 + b1
    ln_leaky_kernel<<<n, 128, 128 * sizeof(float), stream>>>(
        hidb, P(base + 5), P(base + 4), hidb, 128, n, 0);          // LN(g,be)+leaky
    out_head_kernel<<<(n * dout + 255) / 256, 256, 0, stream>>>(
        hidb, P(base + 1), P(base + 3), mask, fixedv, fixld, outt, n, dout);
  };

  auto physics = [&]() {
    outmlp(h_bus, NB, MLPB, mask_bus, x_bus, 16, 4, bus_t);
    outmlp(h_gen, NG, MLPG, mask_gen, x_gen, 8, 2, gen_t);
    zero(aggb, (size_t)NB * 2);
    agg_kernel<<<(NG * 2 + 255) / 256, 256, 0, stream>>>(gen_t, gb_s, gb_d, NG, aggb);
    vrvi_kernel<<<(NB + 255) / 256, 256, 0, stream>>>(bus_t, vrib, NB);
    zero(irii, (size_t)NB * 2);
    pf_edge_kernel<<<(EBB + 255) / 256, 256, 0, stream>>>(ea_bb, bb_s, bb_d, vrib, irii, EBB);
    pf_res_kernel<<<(NB + 255) / 256, 256, 0, stream>>>(bus_t, aggb, vrib, irii, resb, NB);
    phys_kernel<<<(NB * HC + 255) / 256, 256, 0, stream>>>(resb, P(PHYS), P(PHYS + 1), h_bus, NB);
  };

  // ---- input projections ----
  proj_kernel<<<NB, 128, 0, stream>>>(x_bus, 16, P(PRB), P(PRB + 2), P(PRB + 1),
                                      P(PRB + 3), P(PRB + 5), P(PRB + 4), h0b, NB);
  proj_kernel<<<NG, 128, 0, stream>>>(x_gen, 8, P(PRG), P(PRG + 2), P(PRG + 1),
                                      P(PRG + 3), P(PRG + 5), P(PRG + 4), h0g, NG);
  proj_kernel<<<EBB, 128, 0, stream>>>(ea_bb, 2, P(PRE), P(PRE + 2), P(PRE + 1),
                                       P(PRE + 3), P(PRE + 5), P(PRE + 4), e_emb, EBB);

  // ---- layers ----
  for (int l = 0; l < 3; ++l) {
    int L = 3 + 32 * l;                 // bb:+0(10) bg:+10(9) gb:+19(9) ln_bus:+28 ln_gen:+30
    int K = (l == 0) ? HID : HC;
    const float* hb = (l == 0) ? h0b : h_bus;
    const float* hg = (l == 0) ? h0g : h_gen;
    tconv(hb, NB, hb, NB, K, L + 0,  true,  bb_s, bb_d, EBB, mb, 0);  // bb (overwrite)
    tconv(hg, NG, hb, NB, K, L + 19, false, gb_s, gb_d, NG,  mb, 1);  // gb (accumulate)
    tconv(hb, NB, hg, NG, K, L + 10, false, bg_s, bg_d, NG,  mg, 0);  // bg
    ln_leaky_kernel<<<NB, HC, HC * sizeof(float), stream>>>(
        mb, P(L + 29), P(L + 28), h_bus, HC, NB, l > 0);
    ln_leaky_kernel<<<NG, HC, HC * sizeof(float), stream>>>(
        mg, P(L + 31), P(L + 30), h_gen, HC, NG, l > 0);
    physics();
  }

  // ---- final outputs ----
  outmlp(h_bus, NB, MLPB, mask_bus, x_bus, 16, 4, bus_t);
  outmlp(h_gen, NG, MLPG, mask_gen, x_gen, 8, 2, gen_t);
  zero(aggb, (size_t)NB * 2);
  agg_kernel<<<(NG * 2 + 255) / 256, 256, 0, stream>>>(gen_t, gb_s, gb_d, NG, aggb);
  final_out_kernel<<<(NB * 6 + 255) / 256, 256, 0, stream>>>(bus_t, aggb, (float*)d_out, NB);
}
